// MoeMLP_18897856103233
// MI455X (gfx1250) — matrix operand order repaired and checked by reading, not compiled, not run on hardware
//
#include <hip/hip_runtime.h>
#include <math.h>

#define USE_TDM 1   // CDNA5 Tensor Data Mover for weight/H tile staging

// ---------------- problem constants ----------------
#define T_TOK   8192      // B*H*W tokens
#define C_DIM   1024
#define E_NUM   8
#define FFN_DIM 1024
#define HW_SP   4096
#define NB      2
#define P_TOT   16384     // 2*T (top-2 compact rows total)

typedef __bf16 bf16;
typedef bf16           v16bf __attribute__((ext_vector_type(16)));
typedef bf16           v8bf  __attribute__((ext_vector_type(8)));
typedef float          v8f   __attribute__((ext_vector_type(8)));
typedef unsigned short v8us  __attribute__((ext_vector_type(8)));
typedef unsigned int   u32x4 __attribute__((ext_vector_type(4)));
typedef unsigned int   u32x8 __attribute__((ext_vector_type(8)));

// ---------------- workspace layout (bytes) ----------------
#define WS_XH   ((size_t)0)               // bf16 [C][T]      16MB
#define WS_XL   (WS_XH  + 16777216)       // bf16 [C][T]
#define WS_W1H  (WS_XL  + 16777216)       // bf16 [E][FFN][C]
#define WS_W1L  (WS_W1H + 16777216)
#define WS_W3H  (WS_W1L + 16777216)
#define WS_W3L  (WS_W3H + 16777216)
#define WS_W2H  (WS_W3L + 16777216)       // bf16 [E][C][FFN]
#define WS_W2L  (WS_W2H + 16777216)
#define WS_COMB (WS_W2L + 16777216)       // f32 [T][E]
#define WS_CNT  (WS_COMB + 262144)        // int [E]
#define WS_OFF  (WS_CNT + 256)            // int [E+1]
#define WS_TOK  (WS_OFF + 256)            // int [E][T]
#define WS_HH   (WS_TOK + 262144)         // bf16 [FFN][2T] (f-major!)
#define WS_HL   (WS_HH + 33554432)
#define WS_OUT  (WS_HL + 33554432)        // f32 [C][T] (c-major)
#define WS_NEED (WS_OUT + 33554432)       // ~236 MB

// LDS region offsets, G1 (pitch 80B for 32-k row-major W tiles, 272B for K-major X tiles)
#define SW1H 0
#define SW1L 10240
#define SW3H 20480
#define SW3L 30720
#define SXH  40960
#define SXL  49664
#define G1_LDS 58368
// LDS regions, G2
#define SW2H 0
#define SW2L 10240
#define SHH  20480
#define SHL  29184
#define G2_LDS 37888

// ---- fragment loaders: two 16B LDS loads per lane ----
static __device__ inline v16bf ld2(const unsigned char* p, int off2) {
  union { v16bf v; v8bf h[2]; } u;
  u.h[0] = *(const v8bf*)(p);
  u.h[1] = *(const v8bf*)(p + off2);
  return u.v;
}
// A operand (16x32, row-major tile, pitch 80): lane = row, K halves split by lane>>4
static __device__ inline v16bf ld_a(const unsigned char* base, int row, int sel) {
  return ld2(base + row * 80 + sel * 16, 32);
}
// B operand (32x16, K-major tile, pitch 272): lane%16 = N col, elements = K in table order (half by lane>>4)
static __device__ inline v16bf ld_b(const unsigned char* base, int krow, int ncol) {
  v16bf v;
  const int n = ncol + (krow & 15), h = krow >> 4;
#pragma unroll
  for (int s = 0; s < 16; ++s) {
    int k = (s < 8) ? (8 * h + s) : (16 + 8 * h + (s - 8));
    v[s] = *(const bf16*)(base + k * 272 + n * 2);
  }
  return v;
}

static __device__ inline v8f wmma_bf16(v16bf a, v16bf b, v8f c) {
  return __builtin_amdgcn_wmma_f32_16x16x32_bf16(false, a, false, b, (short)0, c, false, false);
}

#if USE_TDM
// ---- Tensor Data Mover: async 2-D tile load, global -> padded LDS tile ----
// D# per CDNA5 ISA ch.10 sec.8: group0 = {count=1, lds_addr, global_addr, type=2},
// group1 = {wg_mask=0, data_size=2B (code 1), pad_enable, pad_interval/amount,
//           tensor_dim0/1 (large: no clip), tile_dim0/1, dim0_stride}.
// pad codes: interval 2<<pi dwords stored, amount pa+1 dwords inserted.
static __device__ inline void tdm_load_2d(unsigned lds_off, const void* g,
                                          unsigned w_elems, unsigned rows,
                                          unsigned stride_elems,
                                          unsigned pi_code, unsigned pa_code) {
  unsigned long long ga = (unsigned long long)g;
  u32x4 g0;
  g0[0] = 1u;                                            // count=1 (valid user D#)
  g0[1] = lds_off;                                       // lds_addr (bytes)
  g0[2] = (unsigned)ga;                                  // global_addr[31:0]
  g0[3] = ((unsigned)(ga >> 32) & 0x01FFFFFFu) | 0x80000000u;  // addr[56:32] | type=2
  const unsigned td0 = 1u << 20, td1 = 1u << 20;         // tensor dims: large
  u32x8 g1;
  g1[0] = (1u << 16) | (1u << 20) | (pi_code << 22) | (pa_code << 25);
  g1[1] = (td0 & 0xFFFFu) << 16;                         // tensor_dim0[15:0]
  g1[2] = (td0 >> 16) | ((td1 & 0xFFFFu) << 16);         // td0[31:16] | td1[15:0]
  g1[3] = (td1 >> 16) | (w_elems << 16);                 // td1[31:16] | tile_dim0
  g1[4] = rows & 0xFFFFu;                                // tile_dim1 | tile_dim2=0
  g1[5] = stride_elems;                                  // tensor_dim0_stride[31:0]
  g1[6] = 0u;                                            // stride0 hi | stride1 lo
  g1[7] = 0u;
  asm volatile("tensor_load_to_lds %0, %1" :: "s"(g0), "s"(g1) : "memory");
}
static __device__ inline void tdm_wait() {
  asm volatile("s_wait_tensorcnt 0x0" ::: "memory");
}
#endif

// ================= kernels =================

__global__ void k_zero(float* __restrict__ outc) {
  size_t n = (size_t)C_DIM * T_TOK;
  for (size_t i = (size_t)blockIdx.x * blockDim.x + threadIdx.x; i < n;
       i += (size_t)gridDim.x * blockDim.x)
    outc[i] = 0.0f;
}

// (B,C,HW) fp32 -> K-major [C][T] bf16 hi/lo (pure coalesced copy: input already c-major)
__global__ void k_cvt_x(const float* __restrict__ hs, bf16* __restrict__ xh, bf16* __restrict__ xl) {
  size_t n = (size_t)NB * C_DIM * HW_SP;
  for (size_t i = (size_t)blockIdx.x * blockDim.x + threadIdx.x; i < n;
       i += (size_t)gridDim.x * blockDim.x) {
    size_t s = i & (HW_SP - 1);
    size_t c = (i >> 12) & (C_DIM - 1);
    size_t b = i >> 22;
    float v = hs[i];
    bf16 h = (bf16)v;
    size_t o = c * T_TOK + b * HW_SP + s;
    xh[o] = h;
    xl[o] = (bf16)(v - (float)h);
  }
}

__global__ void k_cvt_w(const float* __restrict__ w1, const float* __restrict__ w2,
                        const float* __restrict__ w3,
                        bf16* __restrict__ w1h, bf16* __restrict__ w1l,
                        bf16* __restrict__ w3h, bf16* __restrict__ w3l,
                        bf16* __restrict__ w2h, bf16* __restrict__ w2l) {
  size_t n = (size_t)3 * 8388608;  // 3 matrices, each E*FFN*C = 2^23
  for (size_t i = (size_t)blockIdx.x * blockDim.x + threadIdx.x; i < n;
       i += (size_t)gridDim.x * blockDim.x) {
    int m = (int)(i >> 23);
    size_t r = i & (size_t)8388607;
    const float* s = (m == 0) ? w1 : (m == 1) ? w3 : w2;
    bf16* dh = (m == 0) ? w1h : (m == 1) ? w3h : w2h;
    bf16* dl = (m == 0) ? w1l : (m == 1) ? w3l : w2l;
    float v = s[r];
    bf16 h = (bf16)v;
    dh[r] = h;
    dl[r] = (bf16)(v - (float)h);
  }
}

// lane-per-token router: fp32-accurate logits (hi+lo), softmax, top-2, renorm
__global__ void k_router(const bf16* __restrict__ xh, const bf16* __restrict__ xl,
                         const float* __restrict__ gate, float* __restrict__ comb) {
  int t = blockIdx.x * blockDim.x + threadIdx.x;  // 0..8191
  float acc[E_NUM];
#pragma unroll
  for (int e = 0; e < E_NUM; ++e) acc[e] = 0.0f;
  for (int c = 0; c < C_DIM; ++c) {
    size_t o = (size_t)c * T_TOK + t;
    float xv = (float)xh[o] + (float)xl[o];
#pragma unroll
    for (int e = 0; e < E_NUM; ++e) acc[e] += xv * gate[e * C_DIM + c];
  }
  float mx = acc[0];
#pragma unroll
  for (int e = 1; e < E_NUM; ++e) mx = fmaxf(mx, acc[e]);
  float p[E_NUM], s = 0.0f;
#pragma unroll
  for (int e = 0; e < E_NUM; ++e) { p[e] = expf(acc[e] - mx); s += p[e]; }
#pragma unroll
  for (int e = 0; e < E_NUM; ++e) p[e] /= s;
  int j1 = 0;
#pragma unroll
  for (int e = 1; e < E_NUM; ++e) if (p[e] > p[j1]) j1 = e;
  int j2 = (j1 == 0) ? 1 : 0;
#pragma unroll
  for (int e = 0; e < E_NUM; ++e) if (e != j1 && p[e] > p[j2]) j2 = e;
  float den = p[j1] + p[j2];
  float r1 = p[j1] / den, r2 = p[j2] / den;
#pragma unroll
  for (int e = 0; e < E_NUM; ++e)
    comb[(size_t)t * E_NUM + e] = (e == j1) ? r1 : ((e == j2) ? r2 : 0.0f);
}

// deterministic per-expert ordered compaction (block scan, ascending token order)
__global__ void k_compact(const float* __restrict__ comb, int* __restrict__ tok,
                          int* __restrict__ cnt) {
  int e = blockIdx.x, tid = threadIdx.x;
  __shared__ int sb[256];
  __shared__ int srun;
  if (tid == 0) srun = 0;
  __syncthreads();
  for (int base = 0; base < T_TOK; base += 256) {
    int t = base + tid;
    int f = (comb[(size_t)t * E_NUM + e] > 0.0f) ? 1 : 0;
    sb[tid] = f;
    __syncthreads();
    for (int off = 1; off < 256; off <<= 1) {
      int v = (tid >= off) ? sb[tid - off] : 0;
      __syncthreads();
      sb[tid] += v;
      __syncthreads();
    }
    int pos = srun + sb[tid] - 1;
    int tot = sb[255];
    if (f) tok[(size_t)e * T_TOK + pos] = t;
    __syncthreads();
    if (tid == 0) srun += tot;
    __syncthreads();
  }
  if (tid == 0) cnt[e] = srun;
}

__global__ void k_offsets(const int* __restrict__ cnt, int* __restrict__ off) {
  if (blockIdx.x == 0 && threadIdx.x == 0) {
    int a = 0;
    for (int e = 0; e < E_NUM; ++e) { off[e] = a; a += cnt[e]; }
    off[E_NUM] = a;
  }
}

// G1: Ht[f][p] = gelu(x@w1^T) * (x@w3^T) for compacted rows; bf16x3 via hi/lo split.
// A = W (rows = f, K = c), B = X (K-major, cols = gathered tokens).
__launch_bounds__(256, 1)
__global__ void k_g1(const bf16* __restrict__ xh, const bf16* __restrict__ xl,
                     const bf16* __restrict__ w1h, const bf16* __restrict__ w1l,
                     const bf16* __restrict__ w3h, const bf16* __restrict__ w3l,
                     const int* __restrict__ tok, const int* __restrict__ cnt,
                     const int* __restrict__ off,
                     bf16* __restrict__ Hh, bf16* __restrict__ Hl) {
  const int e = blockIdx.z;
  const int c_e = cnt[e];
  const int m0 = blockIdx.y * 128;  // compact-row tile
  if (m0 >= c_e) return;
  const int f0 = blockIdx.x * 128;  // ffn tile
  const int base = off[e];
  const int tid = threadIdx.x;
  const int lane = tid & 31, wid = tid >> 5;
  const int fm = wid & 1, tn = wid >> 1;  // wave tile: 64 f-rows x 32 token-cols

  __shared__ __align__(16) unsigned char sm[G1_LDS];

  // per-thread gathered token indices (persist across all K-steps)
  const int krow = tid >> 3;
  const int j0 = (tid & 7) << 4;
  int toks[16];
#pragma unroll
  for (int j = 0; j < 16; ++j) {
    int i = m0 + j0 + j;
    toks[j] = tok[(size_t)e * T_TOK + (i < c_e ? i : c_e - 1)];
  }

  v8f accg[4][2], accu[4][2];
  v8f z = {0.f, 0.f, 0.f, 0.f, 0.f, 0.f, 0.f, 0.f};
#pragma unroll
  for (int mi = 0; mi < 4; ++mi)
#pragma unroll
    for (int ni = 0; ni < 2; ++ni) { accg[mi][ni] = z; accu[mi][ni] = z; }

  const unsigned short* xhs = (const unsigned short*)xh;
  const unsigned short* xls = (const unsigned short*)xl;

  for (int k0 = 0; k0 < C_DIM; k0 += 32) {
    __syncthreads();
#if USE_TDM
    // wave 0 issues async TDM tile loads (64B rows + 16B LDS pad -> pitch 80)
    if (tid < 32) {
      size_t wbase = ((size_t)e * FFN_DIM + f0) * C_DIM + k0;
      tdm_load_2d((unsigned)(size_t)(sm + SW1H), w1h + wbase, 32, 128, C_DIM, 3, 3);
      tdm_load_2d((unsigned)(size_t)(sm + SW1L), w1l + wbase, 32, 128, C_DIM, 3, 3);
      tdm_load_2d((unsigned)(size_t)(sm + SW3H), w3h + wbase, 32, 128, C_DIM, 3, 3);
      tdm_load_2d((unsigned)(size_t)(sm + SW3L), w3l + wbase, 32, 128, C_DIM, 3, 3);
    }
#else
#pragma unroll
    for (int it = 0; it < 2; ++it) {
      int i = tid + it * 256;
      int r = i >> 2, part = i & 3;
      size_t go = ((size_t)e * FFN_DIM + f0 + r) * C_DIM + k0 + part * 8;
      int lo = r * 80 + part * 16;
      *(v8bf*)(sm + SW1H + lo) = *(const v8bf*)(w1h + go);
      *(v8bf*)(sm + SW1L + lo) = *(const v8bf*)(w1l + go);
      *(v8bf*)(sm + SW3H + lo) = *(const v8bf*)(w3h + go);
      *(v8bf*)(sm + SW3L + lo) = *(const v8bf*)(w3l + go);
    }
#endif
    // stage X tile K-major via token gather (overlaps the TDM transfers)
    {
      const unsigned short* rh = xhs + (size_t)(k0 + krow) * T_TOK;
      const unsigned short* rl = xls + (size_t)(k0 + krow) * T_TOK;
      unsigned short hv[16], lv[16];
#pragma unroll
      for (int j = 0; j < 16; ++j) { hv[j] = rh[toks[j]]; lv[j] = rl[toks[j]]; }
      v8us h0 = {hv[0], hv[1], hv[2], hv[3], hv[4], hv[5], hv[6], hv[7]};
      v8us h1 = {hv[8], hv[9], hv[10], hv[11], hv[12], hv[13], hv[14], hv[15]};
      v8us l0 = {lv[0], lv[1], lv[2], lv[3], lv[4], lv[5], lv[6], lv[7]};
      v8us l1 = {lv[8], lv[9], lv[10], lv[11], lv[12], lv[13], lv[14], lv[15]};
      int xo = krow * 272 + j0 * 2;
      *(v8us*)(sm + SXH + xo) = h0;
      *(v8us*)(sm + SXH + xo + 16) = h1;
      *(v8us*)(sm + SXL + xo) = l0;
      *(v8us*)(sm + SXL + xo + 16) = l1;
    }
#if USE_TDM
    if (tid < 32) tdm_wait();
#endif
    __syncthreads();

    v16bf Bh[2], Bl[2];
#pragma unroll
    for (int ni = 0; ni < 2; ++ni) {
      Bh[ni] = ld_b(sm + SXH, lane, tn * 32 + ni * 16);
      Bl[ni] = ld_b(sm + SXL, lane, tn * 32 + ni * 16);
    }
#pragma unroll
    for (int mat = 0; mat < 2; ++mat) {
      const int wh = mat ? SW3H : SW1H;
      const int wl = mat ? SW3L : SW1L;
      v16bf Ah[4], Al[4];
#pragma unroll
      for (int mi = 0; mi < 4; ++mi) {
        int row = fm * 64 + mi * 16 + (lane & 15);
        Ah[mi] = ld_a(sm + wh, row, lane >> 4);
        Al[mi] = ld_a(sm + wl, row, lane >> 4);
      }
#pragma unroll
      for (int mi = 0; mi < 4; ++mi)
#pragma unroll
        for (int ni = 0; ni < 2; ++ni) {
          v8f c = mat ? accu[mi][ni] : accg[mi][ni];
          c = wmma_bf16(Ah[mi], Bh[ni], c);  // hi*hi
          c = wmma_bf16(Ah[mi], Bl[ni], c);  // hi*lo
          c = wmma_bf16(Al[mi], Bh[ni], c);  // lo*hi
          if (mat) accu[mi][ni] = c; else accg[mi][ni] = c;
        }
    }
  }

  // epilogue: exact GELU(g)*u, store f-major Ht hi/lo (coalesced), mask i>=cnt
#pragma unroll
  for (int ni = 0; ni < 2; ++ni) {
    int i = m0 + tn * 32 + ni * 16 + (lane & 15);
    bool ok = (i < c_e);
    int p = base + i;
#pragma unroll
    for (int mi = 0; mi < 4; ++mi)
#pragma unroll
      for (int v = 0; v < 8; ++v) {
        float g = accg[mi][ni][v];
        float u = accu[mi][ni][v];
        float h = 0.5f * g * (1.0f + erff(g * 0.70710678118654752f)) * u;
        if (ok) {
          int f = f0 + fm * 64 + mi * 16 + v + ((lane >> 4) << 3);
          size_t o = (size_t)f * P_TOT + p;
          bf16 hb = (bf16)h;
          Hh[o] = hb;
          Hl[o] = (bf16)(h - (float)hb);
        }
      }
  }
}

// G2: out[c][t] += combine * (H @ w2^T). A = w2 (rows c, K=f), B = Ht (K-major, coalesced).
__launch_bounds__(256, 1)
__global__ void k_g2(const bf16* __restrict__ Hh, const bf16* __restrict__ Hl,
                     const bf16* __restrict__ w2h, const bf16* __restrict__ w2l,
                     const int* __restrict__ tok, const int* __restrict__ cnt,
                     const int* __restrict__ off, const float* __restrict__ comb,
                     float* __restrict__ outc) {
  const int e = blockIdx.z;
  const int c_e = cnt[e];
  const int m0 = blockIdx.y * 128;
  if (m0 >= c_e) return;
  const int c0 = blockIdx.x * 128;
  const int base = off[e];
  const int tid = threadIdx.x;
  const int lane = tid & 31, wid = tid >> 5;
  const int fm = wid & 1, tn = wid >> 1;

  __shared__ __align__(16) unsigned char sm[G2_LDS];

  v8f acc[4][2];
  v8f z = {0.f, 0.f, 0.f, 0.f, 0.f, 0.f, 0.f, 0.f};
#pragma unroll
  for (int mi = 0; mi < 4; ++mi)
#pragma unroll
    for (int ni = 0; ni < 2; ++ni) acc[mi][ni] = z;

  for (int k0 = 0; k0 < FFN_DIM; k0 += 32) {
    __syncthreads();
#if USE_TDM
    if (tid < 32) {
      size_t wbase = ((size_t)e * C_DIM + c0) * FFN_DIM + k0;
      tdm_load_2d((unsigned)(size_t)(sm + SW2H), w2h + wbase, 32, 128, FFN_DIM, 3, 3);
      tdm_load_2d((unsigned)(size_t)(sm + SW2L), w2l + wbase, 32, 128, FFN_DIM, 3, 3);
      size_t hbase = (size_t)k0 * P_TOT + base + m0;
      tdm_load_2d((unsigned)(size_t)(sm + SHH), Hh + hbase, 128, 32, P_TOT, 5, 3);
      tdm_load_2d((unsigned)(size_t)(sm + SHL), Hl + hbase, 128, 32, P_TOT, 5, 3);
      tdm_wait();
    }
#else
#pragma unroll
    for (int it = 0; it < 2; ++it) {
      int i = tid + it * 256;
      int r = i >> 2, part = i & 3;
      size_t go = ((size_t)e * C_DIM + c0 + r) * FFN_DIM + k0 + part * 8;
      int lo = r * 80 + part * 16;
      *(v8bf*)(sm + SW2H + lo) = *(const v8bf*)(w2h + go);
      *(v8bf*)(sm + SW2L + lo) = *(const v8bf*)(w2l + go);
    }
#pragma unroll
    for (int it = 0; it < 2; ++it) {
      int i = tid + it * 256;
      int r = i >> 4, part = i & 15;
      size_t go = (size_t)(k0 + r) * P_TOT + base + m0 + part * 8;
      int lo = r * 272 + part * 16;
      *(v8bf*)(sm + SHH + lo) = *(const v8bf*)(Hh + go);
      *(v8bf*)(sm + SHL + lo) = *(const v8bf*)(Hl + go);
    }
#endif
    __syncthreads();

    v16bf Bh[2], Bl[2];
#pragma unroll
    for (int ni = 0; ni < 2; ++ni) {
      Bh[ni] = ld_b(sm + SHH, lane, tn * 32 + ni * 16);
      Bl[ni] = ld_b(sm + SHL, lane, tn * 32 + ni * 16);
    }
    v16bf Ah[4], Al[4];
#pragma unroll
    for (int mi = 0; mi < 4; ++mi) {
      int row = fm * 64 + mi * 16 + (lane & 15);
      Ah[mi] = ld_a(sm + SW2H, row, lane >> 4);
      Al[mi] = ld_a(sm + SW2L, row, lane >> 4);
    }
#pragma unroll
    for (int mi = 0; mi < 4; ++mi)
#pragma unroll
      for (int ni = 0; ni < 2; ++ni) {
        v8f c = acc[mi][ni];
        c = wmma_bf16(Ah[mi], Bh[ni], c);
        c = wmma_bf16(Ah[mi], Bl[ni], c);
        c = wmma_bf16(Al[mi], Bh[ni], c);
        acc[mi][ni] = c;
      }
  }

  // epilogue: scale by combine weight, coalesced f32 atomics into c-major out
#pragma unroll
  for (int ni = 0; ni < 2; ++ni) {
    int i = m0 + tn * 32 + ni * 16 + (lane & 15);
    bool ok = (i < c_e);
    int t = tok[(size_t)e * T_TOK + (ok ? i : 0)];
    float wgt = comb[(size_t)t * E_NUM + e];
#pragma unroll
    for (int mi = 0; mi < 4; ++mi)
#pragma unroll
      for (int v = 0; v < 8; ++v) {
        if (ok) {
          int c = c0 + fm * 64 + mi * 16 + v + ((lane >> 4) << 3);
          atomicAdd(&outc[(size_t)c * T_TOK + t], acc[mi][ni][v] * wgt);
        }
      }
  }
}

// c-major [C][T] -> (B,C,H,W); both sides contiguous in s
__global__ void k_out(const float* __restrict__ outc, float* __restrict__ dout) {
  size_t n = (size_t)NB * C_DIM * HW_SP;
  for (size_t i = (size_t)blockIdx.x * blockDim.x + threadIdx.x; i < n;
       i += (size_t)gridDim.x * blockDim.x) {
    size_t s = i & (HW_SP - 1);
    size_t c = (i >> 12) & (C_DIM - 1);
    size_t b = i >> 22;
    dout[i] = outc[c * T_TOK + b * HW_SP + s];
  }
}

// ================= launch =================
extern "C" void kernel_launch(void* const* d_in, const int* in_sizes, int n_in,
                              void* d_out, int out_size, void* d_ws, size_t ws_size,
                              hipStream_t stream) {
  (void)in_sizes; (void)n_in; (void)out_size; (void)ws_size;  // ws need: ~236MB
  const float* hs   = (const float*)d_in[0];
  const float* gate = (const float*)d_in[1];
  const float* w1   = (const float*)d_in[2];
  const float* w2   = (const float*)d_in[3];
  const float* w3   = (const float*)d_in[4];
  unsigned char* ws = (unsigned char*)d_ws;

  bf16* xh  = (bf16*)(ws + WS_XH);
  bf16* xl  = (bf16*)(ws + WS_XL);
  bf16* w1h = (bf16*)(ws + WS_W1H);
  bf16* w1l = (bf16*)(ws + WS_W1L);
  bf16* w3h = (bf16*)(ws + WS_W3H);
  bf16* w3l = (bf16*)(ws + WS_W3L);
  bf16* w2h = (bf16*)(ws + WS_W2H);
  bf16* w2l = (bf16*)(ws + WS_W2L);
  float* comb = (float*)(ws + WS_COMB);
  int* cnt  = (int*)(ws + WS_CNT);
  int* off  = (int*)(ws + WS_OFF);
  int* tok  = (int*)(ws + WS_TOK);
  bf16* Hh  = (bf16*)(ws + WS_HH);
  bf16* Hl  = (bf16*)(ws + WS_HL);
  float* outc = (float*)(ws + WS_OUT);

  k_zero<<<4096, 256, 0, stream>>>(outc);
  k_cvt_x<<<4096, 256, 0, stream>>>(hs, xh, xl);
  k_cvt_w<<<8192, 256, 0, stream>>>(w1, w2, w3, w1h, w1l, w3h, w3l, w2h, w2l);
  k_router<<<32, 256, 0, stream>>>(xh, xl, gate, comb);
  k_compact<<<E_NUM, 256, 0, stream>>>(comb, tok, cnt);
  k_offsets<<<1, 1, 0, stream>>>(cnt, off);
  k_g1<<<dim3(FFN_DIM / 128, T_TOK / 128, E_NUM), 256, 0, stream>>>(
      xh, xl, w1h, w1l, w3h, w3l, tok, cnt, off, Hh, Hl);
  k_g2<<<dim3(C_DIM / 128, T_TOK / 128, E_NUM), 256, 0, stream>>>(
      Hh, Hl, w2h, w2l, tok, cnt, off, comb, outc);
  k_out<<<4096, 256, 0, stream>>>(outc, (float*)d_out);
}
